// MultiLabelMultiHeadLAAT_74629351735738
// MI455X (gfx1250) — matrix operand order repaired and checked by reading, not compiled, not run on hardware
//
#include <hip/hip_runtime.h>

#define B_ 4
#define S_ 2048
#define D_ 512
#define L_ 4096
#define H_ 2
#define NEG_ (-1.0e6f)

typedef __attribute__((ext_vector_type(16))) __bf16        v16bf;
typedef __attribute__((ext_vector_type(8)))  float         v8f;
typedef __attribute__((ext_vector_type(8)))  unsigned int  v8u;
typedef __attribute__((ext_vector_type(4)))  unsigned int  v4u;
typedef __attribute__((ext_vector_type(8)))  int           v8i;
typedef __attribute__((ext_vector_type(4)))  int           v4i;

// ---------- helpers ----------

__device__ __forceinline__ unsigned short f2bf(float f) {
  unsigned u = __float_as_uint(f);
  u += 0x7fffu + ((u >> 16) & 1u);          // round-to-nearest-even
  return (unsigned short)(u >> 16);
}

__device__ __forceinline__ float bf2f(unsigned short u) {
  return __uint_as_float((unsigned)u << 16);
}

__device__ __forceinline__ v8f vzero8() {
  v8f z = {0.f, 0.f, 0.f, 0.f, 0.f, 0.f, 0.f, 0.f};
  return z;
}

__device__ __forceinline__ v16bf pack_frag(uint4 q1, uint4 q2) {
  v8u pk;
  pk[0] = q1.x; pk[1] = q1.y; pk[2] = q1.z; pk[3] = q1.w;
  pk[4] = q2.x; pk[5] = q2.y; pk[6] = q2.z; pk[7] = q2.w;
  return __builtin_bit_cast(v16bf, pk);
}

// A-matrix fragment 16x32 bf16 from row-major storage (ld in elements).
// ISA 7.12.2: lane m (0..15), g = lane>>4: VGPR v(0..3) K=2v+8g, v(4..7) K=16+2(v-4)+8g
// -> two contiguous 16B loads at column offsets 8g and 16+8g.
__device__ __forceinline__ v16bf load_frag_a(const unsigned short* base, int ld, int lane) {
  int m = lane & 15, g = (lane >> 4) & 1;
  const uint4* p1 = (const uint4*)(base + (size_t)m * ld + 8 * g);
  const uint4* p2 = (const uint4*)(base + (size_t)m * ld + 16 + 8 * g);
  return pack_frag(*p1, *p2);
}

// B-matrix fragment 32x16 bf16; element (k,n) lives at base[n*ld + k].
// lane n holds K = 8g..8g+7 and 16+8g..16+8g+7 (same order as A) -> two 16B loads.
__device__ __forceinline__ v16bf load_frag_b(const unsigned short* base, int ld, int lane) {
  int n = lane & 15, g = (lane >> 4) & 1;
  const uint4* p1 = (const uint4*)(base + (size_t)n * ld + 8 * g);
  const uint4* p2 = (const uint4*)(base + (size_t)n * ld + 16 + 8 * g);
  return pack_frag(*p1, *p2);
}

__device__ __forceinline__ v8f wmma_bf16(v16bf a, v16bf b, v8f c) {
  return __builtin_amdgcn_wmma_f32_16x16x32_bf16(false, a, false, b, (short)0, c,
                                                 false, false);
}

// ---------- kernel 0a: f32 -> bf16 convert ----------

__global__ __launch_bounds__(256) void cvt_kernel(const float* __restrict__ in,
                                                  unsigned short* __restrict__ out,
                                                  int n) {
  int i = blockIdx.x * 256 + threadIdx.x;
  if (i < n) out[i] = f2bf(in[i]);
}

// ---------- kernel 0b: h (B,S,D) f32 -> hT (B,D,S) bf16 ----------

__global__ __launch_bounds__(256) void transpose_kernel(const float* __restrict__ h,
                                                        unsigned short* __restrict__ hT) {
  __shared__ unsigned short tile[32][33];
  int b = blockIdx.z;
  int s0 = blockIdx.x * 32, d0 = blockIdx.y * 32;
  int tx = threadIdx.x, ty = threadIdx.y;  // 32 x 8
#pragma unroll
  for (int j = 0; j < 4; ++j) {
    int s = s0 + ty * 4 + j;
    tile[ty * 4 + j][tx] = f2bf(h[((size_t)b * S_ + s) * D_ + d0 + tx]);
  }
  __syncthreads();
#pragma unroll
  for (int j = 0; j < 4; ++j) {
    int d = d0 + ty * 4 + j;
    hT[((size_t)b * D_ + d) * S_ + s0 + tx] = tile[tx][ty * 4 + j];
  }
}

// ---------- kernel 1: z = tanh(h @ W^T + b), bf16 WMMA, z stored bf16 ----------
// grid = B*S/16 blocks; block = 256 (8 waves). Wave w owns output cols [64w,64w+64).

__global__ __launch_bounds__(256) void zgemm_kernel(const unsigned short* __restrict__ hb,
                                                    const unsigned short* __restrict__ wb,
                                                    const float* __restrict__ Wbias,
                                                    unsigned short* __restrict__ zb) {
  int tid = threadIdx.x, lane = tid & 31, wave = tid >> 5;
  int rBase = blockIdx.x * 16;
  int n0 = wave * 64;
  v8f acc[4] = {vzero8(), vzero8(), vzero8(), vzero8()};
  for (int k0 = 0; k0 < D_; k0 += 32) {
    v16bf a = load_frag_a(hb + (size_t)rBase * D_ + k0, D_, lane);
#pragma unroll
    for (int i = 0; i < 4; ++i) {
      v16bf bfr = load_frag_b(wb + (size_t)(n0 + 16 * i) * D_ + k0, D_, lane);
      acc[i] = wmma_bf16(a, bfr, acc[i]);
    }
  }
  int g = lane >> 4, n = lane & 15;
#pragma unroll
  for (int i = 0; i < 4; ++i) {
    int col = n0 + 16 * i + n;
    float bias = Wbias[col];
#pragma unroll
    for (int r = 0; r < 8; ++r) {
      int row = rBase + r + 8 * g;
      zb[(size_t)row * D_ + col] = f2bf(tanhf(acc[i][r] + bias));
    }
  }
}

// ---------- kernel 2: fused label-attention ----------
// One block per (b, tile of 16 labels) = 32 lh rows (two WMMA M-tiles) to double
// arithmetic intensity vs the z/hT L2 streams. Scores kept bf16 in LDS.
// u-tile staged via Tensor Data Mover (tensor_load_to_lds + s_wait_tensorcnt).

__global__ __launch_bounds__(256) void attn_kernel(const unsigned short* __restrict__ zb,
                                                   const unsigned short* __restrict__ hTb,
                                                   const unsigned short* __restrict__ ub,
                                                   const int* __restrict__ wmask,
                                                   const float* __restrict__ final_w,
                                                   const float* __restrict__ final_b,
                                                   float* __restrict__ out) {
  extern __shared__ __align__(16) char smem[];
  unsigned short* sAlpha = (unsigned short*)smem;               // 32*2048 bf16 = 128 KB
  float* red = (float*)(smem + 131072);                         // 256 f32
  unsigned short* uLds = (unsigned short*)(smem + 132096);      // 32*512 bf16 = 32 KB

  int tid = threadIdx.x, lane = tid & 31, wave = tid >> 5;
  int b = blockIdx.x >> 8;        // L/16 = 256 tiles per batch
  int tile = blockIdx.x & 255;
  int lBase = tile * 16;
  int lhBase = tile * 32;         // = lBase * H

  // ---- stage u tile (32 lh rows x 512 bf16) into LDS via TDM (wave 0 only) ----
  if (wave == 0) {
    unsigned ldsAddr = (unsigned)(uintptr_t)(void*)uLds;            // LDS offset = addr[31:0]
    unsigned long long ga = (unsigned long long)(uintptr_t)(ub + (size_t)lhBase * D_);
    // D# group 0: count=1 | lds_addr | global_addr[56:0] | type=2
    v4u g0 = {1u, ldsAddr, (unsigned)(ga & 0xffffffffu),
              (unsigned)((ga >> 32) & 0x01ffffffu) | 0x80000000u};
    // D# group 1: data_size=2B; tensor 512x32; tile 512x32; dim0 stride=512
    v8i g1 = {(int)(1u << 16),        // [17:16] data_size = 1 (2 bytes)
              (int)(512u << 16),      // tensor_dim0[15:0] << 16
              (int)(32u << 16),       // tensor_dim0[31:16]=0 | tensor_dim1[15:0] << 16
              (int)(512u << 16),      // tensor_dim1[31:16]=0 | tile_dim0 << 16
              (int)32,                // tile_dim1 | tile_dim2=0
              (int)512,               // tensor_dim0_stride[31:0]
              0, 0};
    v4i gz4 = {0, 0, 0, 0};
    v8i gz8 = {0, 0, 0, 0, 0, 0, 0, 0};
    __builtin_amdgcn_tensor_load_to_lds(g0, g1, gz4, gz4, gz8, 0);
    __builtin_amdgcn_s_wait_tensorcnt((short)0);
  }
  __syncthreads();

  // ---- phase 1: scores (32 lh x S) -> bf16 LDS; each wave sweeps 16-wide s stripes ----
  for (int sTile = 0; sTile < S_; sTile += 128) {
    int sW = sTile + wave * 16;
    v8f acc0 = vzero8(), acc1 = vzero8();
    for (int k0 = 0; k0 < D_; k0 += 32) {
      // B(k,n) = z[b, sW+n, k0+k]  (row-major z, ld = D) -- shared by both M-tiles
      v16bf bfr = load_frag_b(zb + ((size_t)b * S_ + sW) * D_ + k0, D_, lane);
      v16bf a0 = load_frag_a(uLds + k0, D_, lane);
      v16bf a1 = load_frag_a(uLds + 16 * D_ + k0, D_, lane);
      acc0 = wmma_bf16(a0, bfr, acc0);
      acc1 = wmma_bf16(a1, bfr, acc1);
    }
    int g = lane >> 4, n = lane & 15;
#pragma unroll
    for (int r = 0; r < 8; ++r) {
      sAlpha[(size_t)(r + 8 * g) * S_ + sW + n] = f2bf(acc0[r]);
      sAlpha[(size_t)(16 + r + 8 * g) * S_ + sW + n] = f2bf(acc1[r]);
    }
  }
  __syncthreads();

  // ---- masked softmax over s: 8 threads per lh row (32 rows) ----
  {
    int r = tid >> 3, t = tid & 7;
    unsigned short* row = sAlpha + (size_t)r * S_;
    float mx = -3.0e38f;
    for (int s = t; s < S_; s += 8) {
      float v = wmask[b * S_ + s] ? bf2f(row[s]) : NEG_;
      row[s] = f2bf(v);
      mx = fmaxf(mx, v);
    }
    red[tid] = mx;
    __syncthreads();
    if (t == 0) {
      float m2 = red[r * 8];
      for (int j = 1; j < 8; ++j) m2 = fmaxf(m2, red[r * 8 + j]);
      red[r * 8] = m2;
    }
    __syncthreads();
    float rmax = red[r * 8];
    __syncthreads();
    float sum = 0.f;
    for (int s = t; s < S_; s += 8) {
      float e = __expf(bf2f(row[s]) - rmax);
      row[s] = f2bf(e);
      sum += e;
    }
    red[tid] = sum;
    __syncthreads();
    if (t == 0) {
      float s2 = 0.f;
      for (int j = 0; j < 8; ++j) s2 += red[r * 8 + j];
      red[r * 8] = s2;
    }
    __syncthreads();
    float inv = 1.0f / red[r * 8];
    for (int s = t; s < S_; s += 8) row[s] = f2bf(bf2f(row[s]) * inv);
  }
  __syncthreads();

  // ---- phase 2: m[32 lh][512 d] = alpha @ h (via hT); wave owns 64 d cols ----
  {
    int d0 = wave * 64;
    v8f acc[2][4] = {{vzero8(), vzero8(), vzero8(), vzero8()},
                     {vzero8(), vzero8(), vzero8(), vzero8()}};
    for (int k0 = 0; k0 < S_; k0 += 32) {
      v16bf a0 = load_frag_a(sAlpha + k0, S_, lane);
      v16bf a1 = load_frag_a(sAlpha + (size_t)16 * S_ + k0, S_, lane);
#pragma unroll
      for (int i = 0; i < 4; ++i) {
        // B(k,n) = hT[b, d0+16i+n, k0+k]  (ld = S) -- shared by both M-tiles
        v16bf bfr = load_frag_b(hTb + ((size_t)b * D_ + d0 + 16 * i) * S_ + k0, S_, lane);
        acc[0][i] = wmma_bf16(a0, bfr, acc[0][i]);
        acc[1][i] = wmma_bf16(a1, bfr, acc[1][i]);
      }
    }
    __syncthreads();  // all waves done reading sAlpha before aliasing it as mBuf
    float* mBuf = (float*)smem;  // 32 x 512 f32 = 64 KB (aliases sAlpha region)
    int g = lane >> 4, n = lane & 15;
#pragma unroll
    for (int mt = 0; mt < 2; ++mt)
#pragma unroll
      for (int i = 0; i < 4; ++i)
#pragma unroll
        for (int r = 0; r < 8; ++r)
          mBuf[(size_t)(16 * mt + r + 8 * g) * D_ + d0 + 16 * i + n] = acc[mt][i][r];
  }
  __syncthreads();

  // ---- head max-pool + final per-label dot (16 threads per label) ----
  {
    int li = tid >> 4;  // 0..15 local label
    int t = tid & 15;
    const float* mBuf = (const float*)smem;
    const float* fw = final_w + (size_t)(lBase + li) * D_;
    float accv = 0.f;
    for (int d = t; d < D_; d += 16) {
      float mv = fmaxf(mBuf[(size_t)(2 * li) * D_ + d], mBuf[(size_t)(2 * li + 1) * D_ + d]);
      accv += mv * fw[d];
    }
#pragma unroll
    for (int off = 8; off > 0; off >>= 1) accv += __shfl_down(accv, off, 16);
    if (t == 0) out[(size_t)b * L_ + lBase + li] = accv + final_b[lBase + li];
  }
}

// ---------- launch ----------

extern "C" void kernel_launch(void* const* d_in, const int* in_sizes, int n_in,
                              void* d_out, int out_size, void* d_ws, size_t ws_size,
                              hipStream_t stream) {
  const float* h = (const float*)d_in[0];
  const int* wmask = (const int*)d_in[1];
  const float* W_w = (const float*)d_in[2];
  const float* W_b = (const float*)d_in[3];
  const float* U_w = (const float*)d_in[4];
  const float* final_w = (const float*)d_in[5];
  const float* final_b = (const float*)d_in[6];
  float* out = (float*)d_out;

  char* ws = (char*)d_ws;
  size_t off = 0;
  unsigned short* hb = (unsigned short*)(ws + off);  off += (size_t)B_ * S_ * D_ * 2;  // h bf16
  unsigned short* zb = (unsigned short*)(ws + off);  off += (size_t)B_ * S_ * D_ * 2;  // z bf16
  unsigned short* hTb = (unsigned short*)(ws + off); off += (size_t)B_ * S_ * D_ * 2;  // hT bf16
  unsigned short* ub = (unsigned short*)(ws + off);  off += (size_t)L_ * H_ * D_ * 2;  // U bf16
  unsigned short* wb = (unsigned short*)(ws + off);  off += (size_t)D_ * D_ * 2;       // W bf16
  (void)ws_size; (void)in_sizes; (void)n_in; (void)out_size;

  int nh = B_ * S_ * D_;
  cvt_kernel<<<(nh + 255) / 256, 256, 0, stream>>>(h, hb, nh);
  int nu = L_ * H_ * D_;
  cvt_kernel<<<(nu + 255) / 256, 256, 0, stream>>>(U_w, ub, nu);
  int nw = D_ * D_;
  cvt_kernel<<<(nw + 255) / 256, 256, 0, stream>>>(W_w, wb, nw);

  transpose_kernel<<<dim3(S_ / 32, D_ / 32, B_), dim3(32, 8), 0, stream>>>(h, hTb);

  zgemm_kernel<<<B_ * S_ / 16, 256, 0, stream>>>(hb, wb, W_b, zb);

  size_t smem2 = 131072 + 1024 + 32 * D_ * 2;  // 164864 B < 320 KB WGP LDS
  attn_kernel<<<B_ * (L_ / 16), 256, smem2, stream>>>(zb, hTb, ub, wmask, final_w,
                                                      final_b, out);
}
